// MMAttn_49160195670476
// MI455X (gfx1250) — matrix operand order repaired and checked by reading, not compiled, not run on hardware
//
#include <hip/hip_runtime.h>

// ---------------------------------------------------------------------------
// MM-DiT joint attention for MI455X (gfx1250, wave32, WMMA).
// - All matmuls: v_wmma_f32_16x16x32_bf16 (f32 accumulate).
// - Weights are transposed once at f32->bf16 conversion time, so GEMM reads
//   A and B fragments as contiguous b128 loads with no LDS and no barriers:
//   the ~17MB of bf16 weights live in MI455X's 192MB L2 across their massive
//   reuse (64-256 blocks per column).
// - Attention: max-free softmax (scores provably bounded in [-8, 9] because
//   q,k are RMS-normalized), one row-sum reduction after the K loop.
//   P transposed via LDS; V pre-transposed so all loads are contiguous.
// ---------------------------------------------------------------------------

typedef __attribute__((ext_vector_type(16))) __bf16 v16bf;
typedef __attribute__((ext_vector_type(8)))  __bf16 v8bf;
typedef __attribute__((ext_vector_type(4)))  __bf16 v4bf;
typedef __attribute__((ext_vector_type(8)))  float  v8f;

#define B_   2
#define N1_  2048
#define N2_  128
#define D_   1024
#define H_   16
#define DH_  64
#define T_   2176
#define TPF_ 256
#define D3_  3072
#define SCALE_ 0.125f   // 1/sqrt(64)
#define EPS_ 1e-6f

static __device__ inline v8bf ld8(const __bf16* p) { return *(const v8bf*)p; }

static __device__ inline v16bf mk16(v8bf lo, v8bf hi) {
  v16bf r;
#pragma unroll
  for (int i = 0; i < 8; ++i) { r[i] = lo[i]; r[i + 8] = hi[i]; }
  return r;
}

static __device__ inline v8f wmma_bf16(v16bf a, v16bf b, v8f c) {
  return __builtin_amdgcn_wmma_f32_16x16x32_bf16(false, a, false, b,
                                                 (short)0, c, false, false);
}

// ---------------------------------------------------------------------------
// f32 -> bf16 conversion (vectorized x4), for activations
// ---------------------------------------------------------------------------
__global__ void cvt_bf16_kernel(const float* __restrict__ src,
                                __bf16* __restrict__ dst, int n4) {
  int i = blockIdx.x * blockDim.x + threadIdx.x;
  int stride = gridDim.x * blockDim.x;
  for (; i < n4; i += stride) {
    float4 f = ((const float4*)src)[i];
    v4bf o;
    o[0] = (__bf16)f.x; o[1] = (__bf16)f.y;
    o[2] = (__bf16)f.z; o[3] = (__bf16)f.w;
    ((v4bf*)dst)[i] = o;
  }
}

// ---------------------------------------------------------------------------
// f32 [K,N] -> bf16 transposed [N,K], LDS-tiled (32x32) so both global sides
// stay coalesced. K%32==0, N%32==0.
// ---------------------------------------------------------------------------
__global__ __launch_bounds__(256)
void transpose_cvt_kernel(const float* __restrict__ src,
                          __bf16* __restrict__ dst, int K, int N) {
  __shared__ float tile[32][33];
  const int k0 = blockIdx.x * 32;
  const int n0 = blockIdx.y * 32;
  const int tx = threadIdx.x;        // 0..31
  const int ty = threadIdx.y;        // 0..7
#pragma unroll
  for (int r = 0; r < 32; r += 8)
    tile[r + ty][tx] = src[(size_t)(k0 + r + ty) * N + n0 + tx];
  __syncthreads();
#pragma unroll
  for (int r = 0; r < 32; r += 8)
    dst[(size_t)(n0 + r + ty) * K + k0 + tx] = (__bf16)tile[tx][r + ty];
}

// ---------------------------------------------------------------------------
// GEMM: C[M,N] = A[M,K](bf16) @ W[K,N] + bias[N], with W given TRANSPOSED as
// Bt[N,K] bf16. Block = 128 thr (4 waves); block tile 64(M) x 64(N); wave w
// owns rows m0+w*16 and all four 16-col sub-tiles -> one A-fragment feeds
// 4 WMMAs per 32-K step. All fragment loads contiguous b128; no LDS, no
// barriers -- B reuse is served by the 192MB L2.
// Requires M%64==0, N%64==0, K%32==0.
// ---------------------------------------------------------------------------
template <int OUTF32>
__global__ __launch_bounds__(128)
void gemm_bf16_kernel(const __bf16* __restrict__ A,
                      const __bf16* __restrict__ Bt,
                      const float* __restrict__ bias,
                      float* __restrict__ Cf, __bf16* __restrict__ Cb,
                      int M, int N, int K) {
  const int tid  = threadIdx.x;
  const int lane = tid & 31;
  const int w    = tid >> 5;            // wave 0..3 -> M sub-tile
  const int m0   = blockIdx.x * 64 + w * 16;
  const int n0   = blockIdx.y * 64;

  const int nL   = lane & 15;
  const int half = lane >> 4;
  const int mb   = half * 8;

  const __bf16* Arow = A  + (size_t)(m0 + nL) * K;
  const __bf16* Br0  = Bt + (size_t)(n0 +  0 + nL) * K;
  const __bf16* Br1  = Bt + (size_t)(n0 + 16 + nL) * K;
  const __bf16* Br2  = Bt + (size_t)(n0 + 32 + nL) * K;
  const __bf16* Br3  = Bt + (size_t)(n0 + 48 + nL) * K;

  v8f c0 = {}, c1 = {}, c2 = {}, c3 = {};

  for (int k0 = 0; k0 < K; k0 += 32) {
    const int ka = k0 + half * 16;      // A-fragment K base
    const int kb = k0 + half * 16;      // B-fragment K base
    __builtin_prefetch(Arow + k0 + 64, 0, 3);
    const v16bf afrag = mk16(ld8(Arow + ka), ld8(Arow + ka + 8));
    c0 = wmma_bf16(afrag, mk16(ld8(Br0 + kb), ld8(Br0 + kb + 8)), c0);
    c1 = wmma_bf16(afrag, mk16(ld8(Br1 + kb), ld8(Br1 + kb + 8)), c1);
    c2 = wmma_bf16(afrag, mk16(ld8(Br2 + kb), ld8(Br2 + kb + 8)), c2);
    c3 = wmma_bf16(afrag, mk16(ld8(Br3 + kb), ld8(Br3 + kb + 8)), c3);
  }

#pragma unroll
  for (int j = 0; j < 4; ++j) {
    const v8f& c = (j == 0) ? c0 : (j == 1) ? c1 : (j == 2) ? c2 : c3;
    const int col = n0 + j * 16 + nL;
    const float bcol = bias[col];
#pragma unroll
    for (int i = 0; i < 8; ++i) {
      float v = c[i] + bcol;
      size_t idx = (size_t)(m0 + mb + i) * N + col;
      if (OUTF32) Cf[idx] = v;
      else        Cb[idx] = (__bf16)v;
    }
  }
}

// ---------------------------------------------------------------------------
// QKV split + per-head RMSNorm. One wave per (row, head); 2 dh elems / lane.
// Writes Q,K as [B,H,T,DH] bf16 and V transposed as [B,H,DH,T] bf16.
// ---------------------------------------------------------------------------
__global__ __launch_bounds__(256)
void norm_split_kernel(const __bf16* __restrict__ qkv,
                       const float* __restrict__ gq,
                       const float* __restrict__ gk,
                       __bf16* __restrict__ Qb, __bf16* __restrict__ Kb,
                       __bf16* __restrict__ Vt, int Ns, int t0, int rows) {
  const int gw   = blockIdx.x * 8 + (threadIdx.x >> 5);
  const int lane = threadIdx.x & 31;
  if (gw >= rows * H_) return;

  const int r = gw / H_, h = gw % H_;
  const int b = r / Ns, t = r % Ns, tt = t0 + t;
  const int dh = lane * 2;

  const __bf16* base = qkv + (size_t)r * D3_ + h * DH_ + dh;
  float q0 = (float)base[0],        q1 = (float)base[1];
  float k0 = (float)base[D_],       k1 = (float)base[D_ + 1];
  float v0 = (float)base[2 * D_],   v1 = (float)base[2 * D_ + 1];

  float sq = q0 * q0 + q1 * q1;
  float sk = k0 * k0 + k1 * k1;
#pragma unroll
  for (int m = 1; m < 32; m <<= 1) {
    sq += __shfl_xor(sq, m, 32);
    sk += __shfl_xor(sk, m, 32);
  }
  const float rq = rsqrtf(sq * (1.0f / DH_) + EPS_);
  const float rk = rsqrtf(sk * (1.0f / DH_) + EPS_);

  const size_t bh = (size_t)(b * H_ + h);
  __bf16* qd = Qb + (bh * T_ + tt) * DH_ + dh;
  __bf16* kd = Kb + (bh * T_ + tt) * DH_ + dh;
  qd[0] = (__bf16)(q0 * rq * gq[dh]);
  qd[1] = (__bf16)(q1 * rq * gq[dh + 1]);
  kd[0] = (__bf16)(k0 * rk * gk[dh]);
  kd[1] = (__bf16)(k1 * rk * gk[dh + 1]);

  __bf16* vd = Vt + (bh * DH_ + dh) * (size_t)T_ + tt;
  vd[0]  = (__bf16)v0;
  vd[T_] = (__bf16)v1;
}

// ---------------------------------------------------------------------------
// Attention, max-free softmax. 1 wave per (b, h, 16-query tile).
// Bound argument: rows of Q,K are RMS-normalized (||.||2 = sqrt(64) = 8,
// g == 1), so |q.k| * 0.125 <= 8; with the +1.0 block-causal mask scores lie
// in [-8, 9]; exp() and the <=2176-term row sum fit easily in f32. Hence no
// running max / rescaling: per chunk just exp -> P -> PV WMMAs, and a single
// cross-lane row-sum reduction at the end.
// ---------------------------------------------------------------------------
__global__ __launch_bounds__(32)
void attn_kernel(const __bf16* __restrict__ Q, const __bf16* __restrict__ Km,
                 const __bf16* __restrict__ Vt,
                 __bf16* __restrict__ xo1, __bf16* __restrict__ xo2) {
  __shared__ __bf16 lp[16 * 48];  // P tile [m][k], padded row stride 96B

  const int lane = threadIdx.x;
  const int qt = blockIdx.x, h = blockIdx.y, b = blockIdx.z;
  const int nL = lane & 15, half = lane >> 4;
  const int mb = half * 8;
  const int kaq = half * 16;

  const __bf16* Qh = Q  + ((size_t)(b * H_ + h) * T_) * DH_;
  const __bf16* Kh = Km + ((size_t)(b * H_ + h) * T_) * DH_;
  const __bf16* Vh = Vt + ((size_t)(b * H_ + h) * DH_) * T_;

  const int qtok = qt * 16 + nL;
  const __bf16* Qr = Qh + (size_t)qtok * DH_;
  const v16bf aq0 = mk16(ld8(Qr + kaq),      ld8(Qr + kaq + 8));
  const v16bf aq1 = mk16(ld8(Qr + 32 + kaq), ld8(Qr + 32 + kaq + 8));

  v8f o0 = {}, o1 = {}, o2 = {}, o3 = {};
  float rsum[8];
#pragma unroll
  for (int i = 0; i < 8; ++i) rsum[i] = 0.0f;

  const int qrow0 = qt * 16 + mb;  // row of C element i is qrow0 + i

  for (int kb0 = 0; kb0 < T_; kb0 += 32) {
    __syncthreads();  // previous iteration's ap reads complete (WAR)

    // ---- scores -> exp -> P tile in LDS --------------------------------
#pragma unroll
    for (int sub = 0; sub < 2; ++sub) {
      const int ktok = kb0 + sub * 16 + nL;
      const __bf16* Kr = Kh + (size_t)ktok * DH_ + half * 16;
      v16bf bk0 = mk16(ld8(Kr),      ld8(Kr + 8));
      v16bf bk1 = mk16(ld8(Kr + 32), ld8(Kr + 40));
      v8f t = {};
      t = wmma_bf16(aq0, bk0, t);
      t = wmma_bf16(aq1, bk1, t);
#pragma unroll
      for (int i = 0; i < 8; ++i) {
        const int qr = qrow0 + i;
        const float m = ((qr < N1_) && (ktok < N1_) &&
                         ((ktok >> 8) > (qr >> 8))) ? 1.0f : 0.0f;
        const float p = __expf(t[i] * SCALE_ + m);
        rsum[i] += p;  // per-lane partial (this lane's key column only)
        lp[(mb + i) * 48 + sub * 16 + nL] = (__bf16)p;
      }
    }
    __syncthreads();

    // ---- P as A-fragment, O += P @ V -----------------------------------
    const v16bf ap = mk16(ld8(&lp[nL * 48 + kaq]),
                          ld8(&lp[nL * 48 + kaq + 8]));
    {
      const __bf16* Vr = Vh + (size_t)(0 * 16 + nL) * T_ + kb0 + half * 16;
      o0 = wmma_bf16(ap, mk16(ld8(Vr), ld8(Vr + 8)), o0);
    }
    {
      const __bf16* Vr = Vh + (size_t)(1 * 16 + nL) * T_ + kb0 + half * 16;
      o1 = wmma_bf16(ap, mk16(ld8(Vr), ld8(Vr + 8)), o1);
    }
    {
      const __bf16* Vr = Vh + (size_t)(2 * 16 + nL) * T_ + kb0 + half * 16;
      o2 = wmma_bf16(ap, mk16(ld8(Vr), ld8(Vr + 8)), o2);
    }
    {
      const __bf16* Vr = Vh + (size_t)(3 * 16 + nL) * T_ + kb0 + half * 16;
      o3 = wmma_bf16(ap, mk16(ld8(Vr), ld8(Vr + 8)), o3);
    }
  }

  // ---- one row-sum reduction, normalize, merge heads, split streams ----
#pragma unroll
  for (int i = 0; i < 8; ++i) {
    float rs = rsum[i];
    rs += __shfl_xor(rs, 1, 32);
    rs += __shfl_xor(rs, 2, 32);
    rs += __shfl_xor(rs, 4, 32);
    rs += __shfl_xor(rs, 8, 32);
    const float inv = 1.0f / rs;
    const int tok = qrow0 + i;
    __bf16* dst;
    if (tok < N1_) dst = xo1 + ((size_t)(b * N1_ + tok)) * D_ + h * DH_;
    else           dst = xo2 + ((size_t)(b * N2_ + (tok - N1_))) * D_ + h * DH_;
    dst[0 * 16 + nL] = (__bf16)(o0[i] * inv);
    dst[1 * 16 + nL] = (__bf16)(o1[i] * inv);
    dst[2 * 16 + nL] = (__bf16)(o2[i] * inv);
    dst[3 * 16 + nL] = (__bf16)(o3[i] * inv);
  }
}

// ---------------------------------------------------------------------------
// Host-side launch. Workspace need ~88 MB (bf16 weights, qkv, Q/K/V, x-out).
// ---------------------------------------------------------------------------
extern "C" void kernel_launch(void* const* d_in, const int* in_sizes, int n_in,
                              void* d_out, int out_size, void* d_ws,
                              size_t ws_size, hipStream_t stream) {
  (void)in_sizes; (void)n_in; (void)out_size; (void)ws_size;

  const float* x1    = (const float*)d_in[0];
  const float* x2    = (const float*)d_in[1];
  const float* Wqkv1 = (const float*)d_in[2];
  const float* bqkv1 = (const float*)d_in[3];
  const float* Wqkv2 = (const float*)d_in[4];
  const float* bqkv2 = (const float*)d_in[5];
  const float* Wout1 = (const float*)d_in[6];
  const float* bout1 = (const float*)d_in[7];
  const float* Wout2 = (const float*)d_in[8];
  const float* bout2 = (const float*)d_in[9];
  const float* gq1   = (const float*)d_in[10];
  const float* gk1   = (const float*)d_in[11];
  const float* gq2   = (const float*)d_in[12];
  const float* gk2   = (const float*)d_in[13];
  float* out = (float*)d_out;

  const int rows1 = B_ * N1_;  // 4096
  const int rows2 = B_ * N2_;  // 256

  char* base = (char*)d_ws;
  size_t off = 0;
  auto carve = [&](size_t elems) -> __bf16* {
    __bf16* p = (__bf16*)(base + off);
    off = (off + elems * 2 + 255) & ~(size_t)255;
    return p;
  };

  __bf16* Wqkv1t = carve((size_t)D_ * D3_);   // [3D, D] transposed
  __bf16* Wqkv2t = carve((size_t)D_ * D3_);
  __bf16* Wout1t = carve((size_t)D_ * D_);    // [D, D] transposed
  __bf16* Wout2t = carve((size_t)D_ * D_);
  __bf16* x1b    = carve((size_t)rows1 * D_);
  __bf16* x2b    = carve((size_t)rows2 * D_);
  __bf16* qkv1b  = carve((size_t)rows1 * D3_);
  __bf16* qkv2b  = carve((size_t)rows2 * D3_);
  __bf16* Qb     = carve((size_t)B_ * H_ * T_ * DH_);
  __bf16* Kb     = carve((size_t)B_ * H_ * T_ * DH_);
  __bf16* Vtb    = carve((size_t)B_ * H_ * DH_ * T_);
  __bf16* xo1    = carve((size_t)rows1 * D_);
  __bf16* xo2    = carve((size_t)rows2 * D_);

  auto cvt = [&](const float* s, __bf16* d, size_t n) {
    int n4 = (int)(n / 4);
    int blocks = (n4 + 255) / 256;
    cvt_bf16_kernel<<<blocks, 256, 0, stream>>>(s, d, n4);
  };
  auto tcvt = [&](const float* s, __bf16* d, int K, int N) {
    transpose_cvt_kernel<<<dim3(K / 32, N / 32), dim3(32, 8), 0, stream>>>(
        s, d, K, N);
  };

  // 1) activations f32->bf16; weights f32->bf16 transposed [N,K]
  cvt(x1, x1b, (size_t)rows1 * D_);
  cvt(x2, x2b, (size_t)rows2 * D_);
  tcvt(Wqkv1, Wqkv1t, D_, D3_);
  tcvt(Wqkv2, Wqkv2t, D_, D3_);
  tcvt(Wout1, Wout1t, D_, D_);
  tcvt(Wout2, Wout2t, D_, D_);

  // 2) QKV projections (bf16 out)
  gemm_bf16_kernel<0><<<dim3(rows1 / 64, D3_ / 64), 128, 0, stream>>>(
      x1b, Wqkv1t, bqkv1, nullptr, qkv1b, rows1, D3_, D_);
  gemm_bf16_kernel<0><<<dim3(rows2 / 64, D3_ / 64), 128, 0, stream>>>(
      x2b, Wqkv2t, bqkv2, nullptr, qkv2b, rows2, D3_, D_);

  // 3) split + RMSNorm -> Q,K [B,H,T,DH], V^T [B,H,DH,T]
  norm_split_kernel<<<(rows1 * H_) / 8, 256, 0, stream>>>(
      qkv1b, gq1, gk1, Qb, Kb, Vtb, N1_, 0, rows1);
  norm_split_kernel<<<(rows2 * H_) / 8, 256, 0, stream>>>(
      qkv2b, gq2, gk2, Qb, Kb, Vtb, N2_, N1_, rows2);

  // 4) flash attention over concatenated sequence
  attn_kernel<<<dim3(T_ / 16, H_, B_), 32, 0, stream>>>(Qb, Kb, Vtb, xo1, xo2);

  // 5) output projections (f32 out, straight to d_out)
  gemm_bf16_kernel<1><<<dim3(rows1 / 64, D_ / 64), 128, 0, stream>>>(
      xo1, Wout1t, bout1, out, nullptr, rows1, D_, D_);
  gemm_bf16_kernel<1><<<dim3(rows2 / 64, D_ / 64), 128, 0, stream>>>(
      xo2, Wout2t, bout2, out + (size_t)rows1 * D_, nullptr, rows2, D_, D_);
}
